// CrossAttention_19464791786038
// MI455X (gfx1250) — matrix operand order repaired and checked by reading, not compiled, not run on hardware
//
#include <hip/hip_runtime.h>
#include <hip/hip_bf16.h>

#define Bn   16
#define Cc   256
#define Nn   4096
#define Ll   77
#define Lp   96
#define CTXd 768
#define Gg   32

typedef __bf16 bf16_t;
typedef __attribute__((ext_vector_type(16))) __bf16        v16bf;
typedef __attribute__((ext_vector_type(8)))  float         v8f;
typedef __attribute__((ext_vector_type(4)))  float         v4f;
typedef __attribute__((ext_vector_type(4)))  unsigned int  v4u;

union Frag16 {
    v16bf  v;
    v4u    q[2];
    bf16_t h[16];
};

// ---------------------------------------------------------------------------
// Kernel A: k = k_w @ ctx^T, v = v_w @ ctx^T  (per batch).
// Block = 64 threads: wave0 -> k (stored transposed kT[B][96][C]),
//                     wave1 -> v (stored vB[B][C][96]).  L padded 77->96 w/ 0.
// ---------------------------------------------------------------------------
__global__ __launch_bounds__(64) void kv_proj_kernel(
    const float* __restrict__ ctx, const float* __restrict__ k_w,
    const float* __restrict__ v_w, bf16_t* __restrict__ kT,
    bf16_t* __restrict__ vB)
{
    const int b    = blockIdx.x >> 4;
    const int c0   = (blockIdx.x & 15) * 16;
    const int wave = threadIdx.x >> 5;
    const int lane = threadIdx.x & 31;
    const int half = lane >> 4;
    const int l16  = lane & 15;

    const float* W     = wave ? v_w : k_w;
    const float* wrow  = W + (size_t)(c0 + l16) * CTXd;
    const float* cbase = ctx + (size_t)b * Ll * CTXd;

    v8f acc[5];
#pragma unroll
    for (int t = 0; t < 5; ++t) acc[t] = 0.0f;

    for (int k0 = 0; k0 < CTXd; k0 += 32) {
        Frag16 a;
        {
            const float* p0 = wrow + k0 + 16 * half;
            const float* p1 = wrow + k0 + 8 + 16 * half;
#pragma unroll
            for (int i = 0; i < 8; ++i) {
                a.h[i]     = (bf16_t)p0[i];
                a.h[8 + i] = (bf16_t)p1[i];
            }
        }
#pragma unroll
        for (int t = 0; t < 5; ++t) {
            const int l = t * 16 + l16;
            Frag16 bf;
            if (l < Ll) {
                const float* p = cbase + (size_t)l * CTXd + k0 + 16 * half;
#pragma unroll
                for (int i = 0; i < 16; ++i) bf.h[i] = (bf16_t)p[i];
            } else {
#pragma unroll
                for (int i = 0; i < 16; ++i) bf.h[i] = (bf16_t)0.0f;
            }
            acc[t] = __builtin_amdgcn_wmma_f32_16x16x32_bf16(
                false, a.v, false, bf.v, (short)0, acc[t], false, false);
        }
    }

    if (wave == 0) {
        // kT[b][l][c]; D-frag rows (c) are contiguous -> one 16B store per tile
#pragma unroll
        for (int t = 0; t < 5; ++t) {
            const int l = t * 16 + l16;
            Frag16 o;
#pragma unroll
            for (int r = 0; r < 8; ++r) o.h[r] = (bf16_t)acc[t][r];
            *(v4u*)(kT + ((size_t)b * Lp + l) * Cc + c0 + 8 * half) = o.q[0];
        }
        { // zero pad l = 80..95
            const int l = 80 + l16;
            Frag16 z;
#pragma unroll
            for (int r = 0; r < 8; ++r) z.h[r] = (bf16_t)0.0f;
            *(v4u*)(kT + ((size_t)b * Lp + l) * Cc + c0 + 8 * half) = z.q[0];
        }
    } else {
        // vB[b][c][l]
#pragma unroll
        for (int t = 0; t < 5; ++t) {
            const int l = t * 16 + l16;
#pragma unroll
            for (int r = 0; r < 8; ++r) {
                const int c = c0 + 8 * half + r;
                vB[((size_t)b * Cc + c) * Lp + l] = (bf16_t)acc[t][r];
            }
        }
        {
            const int l = 80 + l16;
#pragma unroll
            for (int r = 0; r < 8; ++r) {
                const int c = c0 + 8 * half + r;
                vB[((size_t)b * Cc + c) * Lp + l] = (bf16_t)0.0f;
            }
        }
    }
}

// ---------------------------------------------------------------------------
// Kernel B: q = q_w @ x + q_b, stored transposed qT[B][N][C] (bf16).
// Block 256 thr (8 waves), tile 128(o) x 64(n), K staged through LDS.
// ---------------------------------------------------------------------------
__global__ __launch_bounds__(256) void qproj_kernel(
    const float* __restrict__ x, const float* __restrict__ q_w,
    const float* __restrict__ q_b, bf16_t* __restrict__ qT)
{
    __shared__ __align__(16) bf16_t As[128 * 32];
    __shared__ __align__(16) bf16_t Bs[64 * 32];   // transposed [n][k]

    const int b     = blockIdx.z;
    const int obase = blockIdx.y * 128;
    const int nbase = blockIdx.x * 64;
    const int tid   = threadIdx.x;
    const int wave  = tid >> 5, lane = tid & 31;
    const int half  = lane >> 4, l16 = lane & 15;

    v8f acc[4];
#pragma unroll
    for (int t = 0; t < 4; ++t) acc[t] = 0.0f;

    const float* xb = x + (size_t)b * Cc * Nn;

    for (int k0 = 0; k0 < Cc; k0 += 32) {
        __syncthreads();
#pragma unroll
        for (int e = 0; e < 16; ++e) {                 // A tile 128x32
            int idx = tid + e * 256;
            int r = idx >> 5, k = idx & 31;
            As[idx] = (bf16_t)q_w[(size_t)(obase + r) * Cc + k0 + k];
        }
#pragma unroll
        for (int e = 0; e < 8; ++e) {                  // B tile 32x64, transposed
            int idx = tid + e * 256;
            int kk = idx >> 6, cc = idx & 63;
            Bs[cc * 32 + kk] = (bf16_t)xb[(size_t)(k0 + kk) * Nn + nbase + cc];
        }
        if (k0 + 32 < Cc)                               // prefetch next x slab
            __builtin_prefetch(xb + (size_t)(k0 + 32 + (tid >> 3)) * Nn + nbase, 0, 0);
        __syncthreads();

        Frag16 a;
        {
            const bf16_t* p = As + (wave * 16 + l16) * 32;
            a.q[0] = *(const v4u*)(p + 16 * half);
            a.q[1] = *(const v4u*)(p + 8 + 16 * half);
        }
#pragma unroll
        for (int t = 0; t < 4; ++t) {
            Frag16 bf;
            const bf16_t* p = Bs + (t * 16 + l16) * 32 + 16 * half;
            bf.q[0] = *(const v4u*)(p);
            bf.q[1] = *(const v4u*)(p + 8);
            acc[t] = __builtin_amdgcn_wmma_f32_16x16x32_bf16(
                false, a.v, false, bf.v, (short)0, acc[t], false, false);
        }
    }

    const int orow = obase + wave * 16 + 8 * half;
    float bias[8];
#pragma unroll
    for (int r = 0; r < 8; ++r) bias[r] = q_b[orow + r];
#pragma unroll
    for (int t = 0; t < 4; ++t) {
        const int n = nbase + t * 16 + l16;
        Frag16 o;
#pragma unroll
        for (int r = 0; r < 8; ++r) o.h[r] = (bf16_t)(acc[t][r] + bias[r]);
        *(v4u*)(qT + ((size_t)b * Nn + n) * Cc + orow) = o.q[0];
    }
}

// ---------------------------------------------------------------------------
// Kernel C: fused attention. scores = q^T k  -> softmax(L) -> h^T = w v^T.
// Block 128 thr (4 waves, 16 n-rows each). k then v staged in same 48KB LDS.
// Scores never touch global memory.
// ---------------------------------------------------------------------------
__global__ __launch_bounds__(128) void attn_kernel(
    const bf16_t* __restrict__ qT, const bf16_t* __restrict__ kT,
    const bf16_t* __restrict__ vB, bf16_t* __restrict__ hB)
{
    __shared__ __align__(16) bf16_t sKV[Lp * Cc];        // 48 KB
    __shared__ __align__(16) bf16_t sW[4][16 * Lp];      // 12 KB

    const int b    = blockIdx.y;
    const int nb   = blockIdx.x * 64;
    const int tid  = threadIdx.x;
    const int wave = tid >> 5, lane = tid & 31;
    const int half = lane >> 4, l16 = lane & 15;
    const int nrow = nb + wave * 16;

    {   // stage kT[b] (96 x 256 bf16)
        const v4u* src = (const v4u*)(kT + (size_t)b * Lp * Cc);
        v4u* dst = (v4u*)sKV;
#pragma unroll
        for (int e = 0; e < 24; ++e) dst[tid + e * 128] = src[tid + e * 128];
    }
    __syncthreads();

    v8f s[5];
#pragma unroll
    for (int t = 0; t < 5; ++t) s[t] = 0.0f;

    const bf16_t* qrow = qT + ((size_t)b * Nn + nrow + l16) * Cc;
    for (int k0 = 0; k0 < Cc; k0 += 32) {
        Frag16 a;
        a.q[0] = *(const v4u*)(qrow + k0 + 16 * half);
        a.q[1] = *(const v4u*)(qrow + k0 + 8 + 16 * half);
#pragma unroll
        for (int t = 0; t < 5; ++t) {
            Frag16 bf;
            const bf16_t* p = sKV + (t * 16 + l16) * Cc + k0 + 16 * half;
            bf.q[0] = *(const v4u*)p;
            bf.q[1] = *(const v4u*)(p + 8);
            s[t] = __builtin_amdgcn_wmma_f32_16x16x32_bf16(
                false, a.v, false, bf.v, (short)0, s[t], false, false);
        }
    }

    // softmax over L (columns spread over 16 lanes x 5 tiles; rows r+8*half)
    float w_[5][8];
#pragma unroll
    for (int r = 0; r < 8; ++r) {
        float mx = -3.4e38f;
#pragma unroll
        for (int t = 0; t < 5; ++t) {
            const int l = t * 16 + l16;
            mx = fmaxf(mx, (l < Ll) ? s[t][r] : -3.4e38f);
        }
#pragma unroll
        for (int m = 1; m <= 8; m <<= 1) mx = fmaxf(mx, __shfl_xor(mx, m, 32));
        float sum = 0.f;
#pragma unroll
        for (int t = 0; t < 5; ++t) {
            const int l = t * 16 + l16;
            float p = (l < Ll) ? __expf(s[t][r] - mx) : 0.f;
            w_[t][r] = p;
            sum += p;
        }
#pragma unroll
        for (int m = 1; m <= 8; m <<= 1) sum += __shfl_xor(sum, m, 32);
        const float inv = 1.0f / sum;
#pragma unroll
        for (int t = 0; t < 5; ++t) w_[t][r] *= inv;
    }

    // park softmax weights in per-wave LDS scratch (re-enter A-frag layout)
#pragma unroll
    for (int r = 0; r < 8; ++r) {
        const int m = r + 8 * half;
#pragma unroll
        for (int t = 0; t < 5; ++t)
            sW[wave][m * Lp + t * 16 + l16] = (bf16_t)w_[t][r];
        sW[wave][m * Lp + 80 + l16] = (bf16_t)0.0f;       // L pad 80..95
    }

    __syncthreads();
    {   // stage vB[b] (256 x 96 bf16) into same LDS
        const v4u* src = (const v4u*)(vB + (size_t)b * Cc * Lp);
        v4u* dst = (v4u*)sKV;
#pragma unroll
        for (int e = 0; e < 24; ++e) dst[tid + e * 128] = src[tid + e * 128];
    }
    __syncthreads();

    Frag16 aw[3];
#pragma unroll
    for (int kt = 0; kt < 3; ++kt) {
        const bf16_t* p = &sW[wave][l16 * Lp + kt * 32];
        aw[kt].q[0] = *(const v4u*)(p + 16 * half);
        aw[kt].q[1] = *(const v4u*)(p + 8 + 16 * half);
    }

    for (int ct = 0; ct < 16; ++ct) {                     // h^T tile 16 x 256
        v8f acc = 0.0f;
#pragma unroll
        for (int kt = 0; kt < 3; ++kt) {
            Frag16 bf;
            const bf16_t* p = sKV + (ct * 16 + l16) * Lp + kt * 32 + 16 * half;
            bf.q[0] = *(const v4u*)p;
            bf.q[1] = *(const v4u*)(p + 8);
            acc = __builtin_amdgcn_wmma_f32_16x16x32_bf16(
                false, aw[kt].v, false, bf.v, (short)0, acc, false, false);
        }
        const int c = ct * 16 + l16;
        Frag16 o;
#pragma unroll
        for (int r = 0; r < 8; ++r) o.h[r] = (bf16_t)acc[r];
        *(v4u*)(hB + ((size_t)b * Cc + c) * Nn + nrow + 8 * half) = o.q[0];
    }
}

// ---------------------------------------------------------------------------
// Kernel D: out = out_w @ h + out_b + x (residual), f32 into d_out.
// ---------------------------------------------------------------------------
__global__ __launch_bounds__(256) void oproj_kernel(
    const bf16_t* __restrict__ hB, const float* __restrict__ out_w,
    const float* __restrict__ out_b, const float* __restrict__ x,
    float* __restrict__ y)
{
    __shared__ __align__(16) bf16_t As[128 * 32];
    __shared__ __align__(16) bf16_t Bs[64 * 32];

    const int b     = blockIdx.z;
    const int obase = blockIdx.y * 128;
    const int nbase = blockIdx.x * 64;
    const int tid   = threadIdx.x;
    const int wave  = tid >> 5, lane = tid & 31;
    const int half  = lane >> 4, l16 = lane & 15;

    v8f acc[4];
#pragma unroll
    for (int t = 0; t < 4; ++t) acc[t] = 0.0f;

    for (int k0 = 0; k0 < Cc; k0 += 32) {
        __syncthreads();
#pragma unroll
        for (int e = 0; e < 16; ++e) {
            int idx = tid + e * 256;
            int r = idx >> 5, k = idx & 31;
            As[idx] = (bf16_t)out_w[(size_t)(obase + r) * Cc + k0 + k];
        }
#pragma unroll
        for (int e = 0; e < 8; ++e) {
            int idx = tid + e * 256;
            int kk = idx >> 6, cc = idx & 63;
            Bs[cc * 32 + kk] = hB[((size_t)b * Cc + k0 + kk) * Nn + nbase + cc];
        }
        if (k0 + 32 < Cc)                               // prefetch next h slab
            __builtin_prefetch(hB + ((size_t)b * Cc + k0 + 32 + (tid >> 3)) * Nn + nbase, 0, 0);
        __syncthreads();

        Frag16 a;
        {
            const bf16_t* p = As + (wave * 16 + l16) * 32;
            a.q[0] = *(const v4u*)(p + 16 * half);
            a.q[1] = *(const v4u*)(p + 8 + 16 * half);
        }
#pragma unroll
        for (int t = 0; t < 4; ++t) {
            Frag16 bf;
            const bf16_t* p = Bs + (t * 16 + l16) * 32 + 16 * half;
            bf.q[0] = *(const v4u*)(p);
            bf.q[1] = *(const v4u*)(p + 8);
            acc[t] = __builtin_amdgcn_wmma_f32_16x16x32_bf16(
                false, a.v, false, bf.v, (short)0, acc[t], false, false);
        }
    }

    const int orow = obase + wave * 16 + 8 * half;
#pragma unroll
    for (int t = 0; t < 4; ++t) {
        const int n = nbase + t * 16 + l16;
#pragma unroll
        for (int r = 0; r < 8; ++r) {
            const size_t idx = ((size_t)b * Cc + orow + r) * Nn + n;
            y[idx] = acc[t][r] + out_b[orow + r] + x[idx];
        }
    }
}

// ---------------------------------------------------------------------------
// Kernel E: GroupNorm stats — one block per (b,g); 32768 contiguous floats.
// ---------------------------------------------------------------------------
__global__ __launch_bounds__(256) void gnstats_kernel(
    const float* __restrict__ y, float* __restrict__ stats)
{
    __shared__ float red[256];
    const int bg  = blockIdx.x;
    const int tid = threadIdx.x;
    const v4f* p  = (const v4f*)(y + (size_t)bg * (Cc / Gg) * Nn);

    float s = 0.f, s2 = 0.f;
    for (int i = tid; i < 8192; i += 256) {
        v4f v = p[i];
        s  += v.x + v.y + v.z + v.w;
        s2 += v.x * v.x + v.y * v.y + v.z * v.z + v.w * v.w;
    }
    red[tid] = s; __syncthreads();
    for (int off = 128; off > 0; off >>= 1) {
        if (tid < off) red[tid] += red[tid + off];
        __syncthreads();
    }
    const float tot = red[0]; __syncthreads();
    red[tid] = s2; __syncthreads();
    for (int off = 128; off > 0; off >>= 1) {
        if (tid < off) red[tid] += red[tid + off];
        __syncthreads();
    }
    if (tid == 0) {
        const float mean = tot * (1.0f / 32768.0f);
        const float var  = red[0] * (1.0f / 32768.0f) - mean * mean;
        stats[bg * 2]     = mean;
        stats[bg * 2 + 1] = rsqrtf(var + 1e-5f);
    }
}

// ---------------------------------------------------------------------------
// Kernel F: in-place normalize + affine + swish.
// ---------------------------------------------------------------------------
__global__ __launch_bounds__(256) void gnswish_kernel(
    float* __restrict__ y, const float* __restrict__ stats,
    const float* __restrict__ gamma, const float* __restrict__ beta)
{
    const int    i4 = blockIdx.x * 256 + threadIdx.x;
    const size_t i  = (size_t)i4 * 4;
    const int c  = (int)((i >> 12) & 255);
    const int b  = (int)(i >> 20);
    const int bg = b * Gg + (c >> 3);
    const float mean = stats[bg * 2], rstd = stats[bg * 2 + 1];
    const float ga = gamma[c] * rstd;
    const float be = beta[c] - mean * ga;
    v4f v = *(v4f*)(y + i);
#pragma unroll
    for (int j = 0; j < 4; ++j) {
        const float hn = v[j] * ga + be;
        v[j] = hn / (1.0f + __expf(-hn));
    }
    *(v4f*)(y + i) = v;
}

// ---------------------------------------------------------------------------
extern "C" void kernel_launch(void* const* d_in, const int* in_sizes, int n_in,
                              void* d_out, int out_size, void* d_ws, size_t ws_size,
                              hipStream_t stream)
{
    (void)in_sizes; (void)n_in; (void)out_size; (void)ws_size;
    const float* x     = (const float*)d_in[0];
    const float* ctx   = (const float*)d_in[1];
    const float* q_w   = (const float*)d_in[2];
    const float* q_b   = (const float*)d_in[3];
    const float* k_w   = (const float*)d_in[4];
    const float* v_w   = (const float*)d_in[5];
    const float* out_w = (const float*)d_in[6];
    const float* out_b = (const float*)d_in[7];
    const float* gamma = (const float*)d_in[8];
    const float* beta  = (const float*)d_in[9];
    float* y = (float*)d_out;

    bf16_t* kT = (bf16_t*)d_ws;                          // [B][96][C]
    bf16_t* vB = kT + (size_t)Bn * Lp * Cc;              // [B][C][96]
    bf16_t* qT = vB + (size_t)Bn * Lp * Cc;              // [B][N][C]
    bf16_t* hB = qT + (size_t)Bn * Nn * Cc;              // [B][C][N]
    float* stats = (float*)(hB + (size_t)Bn * Cc * Nn);  // [B*G][2]

    kv_proj_kernel<<<dim3(Bn * 16), 64, 0, stream>>>(ctx, k_w, v_w, kT, vB);
    qproj_kernel<<<dim3(Nn / 64, Cc / 128, Bn), 256, 0, stream>>>(x, q_w, q_b, qT);
    attn_kernel<<<dim3(Nn / 64, Bn), 128, 0, stream>>>(qT, kT, vB, hB);
    oproj_kernel<<<dim3(Nn / 64, Cc / 128, Bn), 256, 0, stream>>>(hB, out_w, out_b, x, y);
    gnstats_kernel<<<dim3(Bn * Gg), 256, 0, stream>>>(y, stats);
    gnswish_kernel<<<dim3((Bn * Cc * Nn) / 4 / 256), 256, 0, stream>>>(y, stats, gamma, beta);
}
